// NodewiseInteraction_29154238005850
// MI455X (gfx1250) — compile-verified, matrix operand order checked by reading, not run on hardware
//
#include <hip/hip_runtime.h>
#include <hip/hip_bf16.h>

#define N_NODES 10000
#define N_EDGES 160000
#define C       128
#define RBF     20
#define HID     32
#define WN      384    // 3*C
#define MSG     1152   // 9*C

typedef __attribute__((ext_vector_type(16))) __bf16       bf16x16;
typedef __attribute__((ext_vector_type(8)))  __bf16       bf16x8;
typedef __attribute__((ext_vector_type(8)))  float        f32x8;
typedef __attribute__((ext_vector_type(4)))  unsigned int u32x4;
typedef __attribute__((ext_vector_type(8)))  int          i32x8;
typedef __attribute__((ext_vector_type(4)))  int          i32x4;

// K index held by vector element i for a 16-bit A/B fragment of 16x16x32 WMMA.
// lanes 0-15 (half=0): elems 0..7 -> K0..7,  elems 8..15 -> K16..23
// lanes 16-31(half=1): elems 0..7 -> K8..15, elems 8..15 -> K24..31
// => per lane: two contiguous runs of 8, starting at 8*half and 16+8*half.
__device__ __forceinline__ int kmap16(int i, int half) {
    return ((i < 8) ? i : (i + 8)) + 8 * half;
}

__device__ __forceinline__ float silu_f(float x) {
    return x / (1.0f + __expf(-x));
}

__device__ __forceinline__ f32x8 wmma_bf16(bf16x16 a, bf16x16 b, f32x8 c) {
    return __builtin_amdgcn_wmma_f32_16x16x32_bf16(false, a, false, b, (short)0, c,
                                                   false, false);
}

// Two 16-byte loads of pre-converted bf16 -> one fragment.
__device__ __forceinline__ bf16x16 frag_from_bf16(const __bf16* p0, const __bf16* p1) {
    bf16x8 lo = *(const bf16x8*)p0;
    bf16x8 hi = *(const bf16x8*)p1;
    return __builtin_shufflevector(lo, hi, 0, 1, 2, 3, 4, 5, 6, 7,
                                           8, 9, 10, 11, 12, 13, 14, 15);
}

// 8 contiguous f32 (32B aligned) -> bf16x8 (packed cvt).
__device__ __forceinline__ bf16x8 cvt8(const float* p) {
    float4 u = *(const float4*)p;
    float4 v = *(const float4*)(p + 4);
    bf16x8 r;
    r[0] = (__bf16)u.x; r[1] = (__bf16)u.y; r[2] = (__bf16)u.z; r[3] = (__bf16)u.w;
    r[4] = (__bf16)v.x; r[5] = (__bf16)v.y; r[6] = (__bf16)v.z; r[7] = (__bf16)v.w;
    return r;
}

__device__ __forceinline__ bf16x16 frag_from_f32(const float* p0, const float* p1) {
    bf16x8 lo = cvt8(p0);
    bf16x8 hi = cvt8(p1);
    return __builtin_shufflevector(lo, hi, 0, 1, 2, 3, 4, 5, 6, 7,
                                           8, 9, 10, 11, 12, 13, 14, 15);
}

// 1-D TDM descriptor: copy nelem bf16 (2B each) from global -> LDS.
// This toolchain exposes the 6-arg builtin: (g0, g1, g2, g3, g4, cpol).
__device__ __forceinline__ void tdm_load_1d(unsigned lds_off, const void* gptr,
                                            unsigned nelem) {
    unsigned long long ga = (unsigned long long)gptr;
    u32x4 g0;
    g0[0] = 1u;                                            // count=1 (valid user D#)
    g0[1] = lds_off;                                       // lds_addr
    g0[2] = (unsigned)(ga & 0xFFFFFFFFu);                  // global_addr[31:0]
    g0[3] = (unsigned)((ga >> 32) & 0x01FFFFFFu) | (2u << 30); // addr[56:32], type=2
    i32x8 g1;
    g1[0] = (int)(1u << 16);                               // wg_mask=0, data_size=1 (2B)
    g1[1] = (int)((nelem & 0xFFFFu) << 16);                // tensor_dim0[15:0]
    g1[2] = (int)(((nelem >> 16) & 0xFFFFu) | (1u << 16)); // tensor_dim0[31:16], dim1=1
    g1[3] = (int)((nelem & 0xFFFFu) << 16);                // tile_dim0 = nelem
    g1[4] = 1;                                             // tile_dim1 = 1
    g1[5] = (int)nelem;                                    // tensor_dim0_stride[31:0]
    g1[6] = 0;
    g1[7] = 0;
    i32x4 g2 = {0, 0, 0, 0};
    i32x4 g3 = {0, 0, 0, 0};
    i32x8 g4 = {0, 0, 0, 0, 0, 0, 0, 0};
    __builtin_amdgcn_tensor_load_to_lds(g0, g1, g2, g3, g4, 0);
}

__global__ void zero_kernel(float* __restrict__ p, size_t n) {
    size_t i   = (size_t)blockIdx.x * blockDim.x + threadIdx.x;
    size_t str = (size_t)gridDim.x * blockDim.x;
    for (; i < n; i += str) p[i] = 0.0f;
}

// Convert all weights to transposed bf16 (Wt[n][k]) so fragment loads are b128.
__global__ void prep_weights(
    const float* __restrict__ Ws1, const float* __restrict__ Wr1,
    const float* __restrict__ Ws2, const float* __restrict__ Wr2,
    const float* __restrict__ w0,  const float* __restrict__ w1,
    const float* __restrict__ w2,
    __bf16* __restrict__ ws1t, __bf16* __restrict__ wr1t,
    __bf16* __restrict__ ws2t, __bf16* __restrict__ wr2t,
    __bf16* __restrict__ w0t,  __bf16* __restrict__ w1t, __bf16* __restrict__ w2t)
{
    const int tid  = blockIdx.x * blockDim.x + threadIdx.x;
    const int nthr = gridDim.x * blockDim.x;
    // ws1t[n][k], n<HID, k<2C
    for (int i = tid; i < HID * 2 * C; i += nthr) {
        int n = i / (2 * C), k = i % (2 * C);
        ws1t[i] = (__bf16)Ws1[k * HID + n];
    }
    // wr1t[n][k], n<HID, k<HID (zero-padded beyond RBF)
    for (int i = tid; i < HID * HID; i += nthr) {
        int n = i / HID, k = i % HID;
        wr1t[i] = (__bf16)((k < RBF) ? Wr1[k * HID + n] : 0.0f);
    }
    // ws2t/wr2t[n][k], n<WN, k<HID
    for (int i = tid; i < WN * HID; i += nthr) {
        int n = i / HID, k = i % HID;
        ws2t[i] = (__bf16)Ws2[k * WN + n];
        wr2t[i] = (__bf16)Wr2[k * WN + n];
    }
    // w*t[d][c], d<C, c<C
    for (int i = tid; i < C * C; i += nthr) {
        int d = i / C, c = i % C;
        w0t[i] = (__bf16)w0[c * C + d];
        w1t[i] = (__bf16)w1[c * C + d];
        w2t[i] = (__bf16)w2[c * C + d];
    }
}

// One wave32 per 16-edge tile; 8 waves/block. Weight panels are TDM-staged
// into LDS as bf16 once per block; both MLPs run on bf16 WMMA with f32 acc;
// messages are scattered with f32 atomics into the L2-resident accumulator.
__global__ __launch_bounds__(256) void edge_kernel(
    const float* __restrict__ node_feat,
    const float* __restrict__ edge_attr,
    const float* __restrict__ edge_rsh,
    const int*   __restrict__ edge_index,
    const __bf16* __restrict__ ws1t, const float* __restrict__ bs1,
    const __bf16* __restrict__ ws2t, const float* __restrict__ bs2,
    const __bf16* __restrict__ wr1t, const float* __restrict__ br1,
    const __bf16* __restrict__ wr2t, const float* __restrict__ br2,
    float* __restrict__ accu)
{
    __shared__ __bf16 l_ws1t[HID * 2 * C];   // 16 KB
    __shared__ __bf16 l_wr1t[HID * HID];     //  2 KB
    __shared__ __bf16 l_ws2t[WN * HID];      // 24 KB
    __shared__ __bf16 l_wr2t[WN * HID];      // 24 KB
    __shared__ float  h_s[8][16][HID];       // 16 KB
    __shared__ float  h_r[8][16][HID];       // 16 KB

    const int wave = threadIdx.x >> 5;
    const int lane = threadIdx.x & 31;
    const int half = lane >> 4;
    const int row  = lane & 15;
    const int tile = blockIdx.x * 8 + wave;
    const int e0   = tile * 16;

    // ---- stage weight panels into LDS via Tensor Data Mover ----
    if (wave == 0) {
        tdm_load_1d((unsigned)(unsigned long long)(void*)l_ws1t, ws1t, HID * 2 * C);
        tdm_load_1d((unsigned)(unsigned long long)(void*)l_wr1t, wr1t, HID * HID);
        tdm_load_1d((unsigned)(unsigned long long)(void*)l_ws2t, ws2t, WN * HID);
        tdm_load_1d((unsigned)(unsigned long long)(void*)l_wr2t, wr2t, WN * HID);
        __builtin_amdgcn_s_wait_tensorcnt(0);
    }
    __syncthreads();

    const int eA   = e0 + row;
    const int srcA = edge_index[eA];
    const int dstA = edge_index[N_EDGES + eA];
    const float* xrow_s = node_feat + (size_t)srcA * C;
    const float* xrow_d = node_feat + (size_t)dstA * C;

    // ---- hidden_s = silu([xi|xj] @ Ws1 + bs1): 16x256 @ 256x32 ----
    f32x8 acc0 = {};
    f32x8 acc1 = {};
    for (int kb = 0; kb < 8; ++kb) {
        const int k0 = kb * 32 + 8 * half;       // run starts (8 floats each)
        const int k1 = k0 + 16;
        const float* p0 = (k0 < C) ? xrow_s + k0 : xrow_d + (k0 - C);
        const float* p1 = (k1 < C) ? xrow_s + k1 : xrow_d + (k1 - C);
        bf16x16 a  = frag_from_f32(p0, p1);
        const __bf16* w0p = l_ws1t + row * (2 * C) + k0;
        const __bf16* w1p = l_ws1t + (16 + row) * (2 * C) + k0;
        bf16x16 b0 = frag_from_bf16(w0p, w0p + 16);
        bf16x16 b1 = frag_from_bf16(w1p, w1p + 16);
        acc0 = wmma_bf16(a, b0, acc0);
        acc1 = wmma_bf16(a, b1, acc1);
    }
    #pragma unroll
    for (int r = 0; r < 8; ++r) {
        int M = r + 8 * half;
        h_s[wave][M][row]      = silu_f(acc0[r] + bs1[row]);
        h_s[wave][M][16 + row] = silu_f(acc1[r] + bs1[16 + row]);
    }

    // ---- hidden_r = silu(edge_attr @ Wr1 + br1): 16x20(pad 32) @ 20x32 ----
    {
        f32x8 r0 = {};
        f32x8 r1 = {};
        bf16x16 a, b0, b1;
        #pragma unroll
        for (int i = 0; i < 16; ++i) {
            int kk = kmap16(i, half);
            a[i] = (__bf16)((kk < RBF) ? edge_attr[(size_t)eA * RBF + kk] : 0.0f);
        }
        const __bf16* w0p = l_wr1t + row * HID + 8 * half;
        const __bf16* w1p = l_wr1t + (16 + row) * HID + 8 * half;
        b0 = frag_from_bf16(w0p, w0p + 16);
        b1 = frag_from_bf16(w1p, w1p + 16);
        r0 = wmma_bf16(a, b0, r0);
        r1 = wmma_bf16(a, b1, r1);
        #pragma unroll
        for (int r = 0; r < 8; ++r) {
            int M = r + 8 * half;
            h_r[wave][M][row]      = silu_f(r0[r] + br1[row]);
            h_r[wave][M][16 + row] = silu_f(r1[r] + br1[16 + row]);
        }
    }

    // LDS h-tiles are private to this wave; wait for DS stores, stop reordering.
    asm volatile("s_wait_dscnt 0" ::: "memory");
    __builtin_amdgcn_wave_barrier();

    const float* hs_row = &h_s[wave][row][0];
    const float* hr_row = &h_r[wave][row][0];
    const int    ro0    = 8 * half;
    const int    ro1    = 16 + 8 * half;

    // ---- second layers + combine + scatter: 24 tiles of 16 output cols ----
    for (int t = 0; t < 24; ++t) {
        bf16x16 ah = frag_from_f32(hs_row + ro0, hs_row + ro1);
        bf16x16 ar = frag_from_f32(hr_row + ro0, hr_row + ro1);
        const int col = t * 16 + row;             // 0..383
        const __bf16* bsp = l_ws2t + col * HID + ro0;
        const __bf16* brp = l_wr2t + col * HID + ro0;
        bf16x16 bh  = frag_from_bf16(bsp, bsp + 16);
        bf16x16 bgr = frag_from_bf16(brp, brp + 16);
        f32x8 z = {};
        f32x8 cs = wmma_bf16(ah, bh, z);
        f32x8 cr = wmma_bf16(ar, bgr, z);

        const int   c     = col & (C - 1);
        const int   which = col >> 7;             // 0 -> w0, 1 -> w1, 2 -> w2
        const float b2s   = bs2[col];
        const float b2r   = br2[col];

        #pragma unroll
        for (int r = 0; r < 8; ++r) {
            int   M  = r + 8 * half;
            int   e  = e0 + M;
            float tp = (cs[r] + b2s) * (cr[r] + b2r);
            int   s  = edge_index[e];
            int   d  = edge_index[N_EDGES + e];
            float wx = tp * node_feat[(size_t)d * C + c];
            float* arow = accu + (size_t)s * MSG;
            if (which == 0) {
                atomicAdd(arow + c, wx * edge_rsh[e * 9 + 0]);
            } else if (which == 1) {
                #pragma unroll
                for (int m = 0; m < 3; ++m)
                    atomicAdd(arow + C + c * 3 + m, wx * edge_rsh[e * 9 + 1 + m]);
            } else {
                #pragma unroll
                for (int m = 0; m < 5; ++m)
                    atomicAdd(arow + 4 * C + c * 5 + m, wx * edge_rsh[e * 9 + 4 + m]);
            }
        }
    }
}

// One wave32 per (16-node tile, gemm id). B from transposed bf16 weights.
__global__ __launch_bounds__(256) void node_kernel(
    const float*  __restrict__ accu,
    const __bf16* __restrict__ w0t, const float* __restrict__ b0,
    const __bf16* __restrict__ w1t, const __bf16* __restrict__ w2t,
    float* __restrict__ out)
{
    const int wave = threadIdx.x >> 5;
    const int lane = threadIdx.x & 31;
    const int half = lane >> 4;
    const int row  = lane & 15;
    const int tile = blockIdx.x * 8 + wave;
    if (tile >= N_NODES / 16) return;
    const int n0 = tile * 16;
    const int g  = blockIdx.y;

    int base, stride, m;
    const __bf16* W;
    if (g == 0)      { base = 0;     stride = 1; m = 0;     W = w0t; }
    else if (g <= 3) { base = C;     stride = 3; m = g - 1; W = w1t; }
    else             { base = 4 * C; stride = 5; m = g - 4; W = w2t; }

    // A fragments for full K=128 (4 k-blocks), reused across 8 output tiles.
    const float* arow_p = accu + (size_t)(n0 + row) * MSG + base + m;
    bf16x16 a[4];
    #pragma unroll
    for (int kb = 0; kb < 4; ++kb) {
        #pragma unroll
        for (int i = 0; i < 16; ++i) {
            int kk = kmap16(i, half) + kb * 32;
            a[kb][i] = (__bf16)arow_p[kk * stride];
        }
    }

    const float inv = 0.08838834764831845f;  // 1/sqrt(C)

    for (int nt = 0; nt < 8; ++nt) {
        const int d = nt * 16 + row;
        f32x8 acc = {};
        #pragma unroll
        for (int kb = 0; kb < 4; ++kb) {
            const __bf16* wp = W + d * C + kb * 32 + 8 * half;
            bf16x16 b = frag_from_bf16(wp, wp + 16);
            acc = wmma_bf16(a[kb], b, acc);
        }
        float bias = (g == 0) ? b0[d] : 0.0f;
        #pragma unroll
        for (int r = 0; r < 8; ++r) {
            int node = n0 + r + 8 * half;
            out[(size_t)node * MSG + base + d * stride + m] = acc[r] * inv + bias;
        }
    }
}

extern "C" void kernel_launch(void* const* d_in, const int* in_sizes, int n_in,
                              void* d_out, int out_size, void* d_ws, size_t ws_size,
                              hipStream_t stream) {
    const float* node_feat  = (const float*)d_in[0];
    const float* edge_attr  = (const float*)d_in[1];
    const float* edge_rsh   = (const float*)d_in[2];
    const int*   edge_index = (const int*)  d_in[3];
    const float* Ws1  = (const float*)d_in[4];
    const float* bs1  = (const float*)d_in[5];
    const float* Ws2  = (const float*)d_in[6];
    const float* bs2  = (const float*)d_in[7];
    const float* Wr1  = (const float*)d_in[8];
    const float* br1  = (const float*)d_in[9];
    const float* Wr2  = (const float*)d_in[10];
    const float* br2  = (const float*)d_in[11];
    const float* l2w0 = (const float*)d_in[12];
    const float* l2b0 = (const float*)d_in[13];
    const float* l2w1 = (const float*)d_in[14];
    const float* l2w2 = (const float*)d_in[15];

    // Workspace layout: accu (f32) then transposed bf16 weights (16B aligned).
    char*  ws   = (char*)d_ws;
    float* accu = (float*)ws;
    size_t off  = (size_t)N_NODES * MSG * sizeof(float);   // 46,080,000 (16B-mult)
    __bf16* ws1t = (__bf16*)(ws + off); off += (size_t)HID * 2 * C * 2;
    __bf16* wr1t = (__bf16*)(ws + off); off += (size_t)HID * HID * 2;
    __bf16* ws2t = (__bf16*)(ws + off); off += (size_t)WN * HID * 2;
    __bf16* wr2t = (__bf16*)(ws + off); off += (size_t)WN * HID * 2;
    __bf16* w0t  = (__bf16*)(ws + off); off += (size_t)C * C * 2;
    __bf16* w1t  = (__bf16*)(ws + off); off += (size_t)C * C * 2;
    __bf16* w2t  = (__bf16*)(ws + off); off += (size_t)C * C * 2;

    zero_kernel<<<512, 256, 0, stream>>>(accu, (size_t)N_NODES * MSG);
    prep_weights<<<128, 256, 0, stream>>>(Ws1, Wr1, Ws2, Wr2, l2w0, l2w1, l2w2,
                                          ws1t, wr1t, ws2t, wr2t, w0t, w1t, w2t);

    edge_kernel<<<N_EDGES / (16 * 8), 256, 0, stream>>>(
        node_feat, edge_attr, edge_rsh, edge_index,
        ws1t, bs1, ws2t, bs2, wr1t, br1, wr2t, br2, accu);

    dim3 ggrid((N_NODES / 16 + 7) / 8, 9);
    node_kernel<<<ggrid, 256, 0, stream>>>(accu, w0t, l2b0, w1t, w2t, (float*)d_out);
}
